// SWABlock_52613349376458
// MI455X (gfx1250) — compile-verified, matrix operand order checked by reading, not run on hardware
//
#include <hip/hip_runtime.h>
#include <hip/hip_bf16.h>
#include <cstdint>
#include <cstddef>

// ---------------------------------------------------------------------------
// CDNA5 (gfx1250) fused Swin window-attention block.
// One workgroup (256 threads = 8 wave32) per 8x16 window.
//  - Window gather: Tensor Data Mover (tensor_load_to_lds) with hardware LDS
//    padding -> conflict-free transposed residual buffer sXT[c][token].
//  - All GEMMs: v_wmma_f32_16x16x32_bf16, f32 accumulate.
//  - Residual stream + every intermediate lives in LDS (~315 KB of 320 KB).
// ---------------------------------------------------------------------------

typedef __bf16 bf16_t;
typedef __attribute__((ext_vector_type(16))) __bf16 bf16x16;
typedef __attribute__((ext_vector_type(8)))  float  f32x8;
typedef __attribute__((ext_vector_type(4)))  unsigned int u32x4;
typedef __attribute__((ext_vector_type(8)))  int i32x8;
typedef __attribute__((ext_vector_type(4)))  int i32x4;

static __device__ __forceinline__ f32x8 wmma_bf16(bf16x16 a, bf16x16 b, f32x8 c) {
    return __builtin_amdgcn_wmma_f32_16x16x32_bf16(false, a, false, b, (short)0, c, false, false);
}

static __device__ __forceinline__ f32x8 zero8() {
    f32x8 z;
#pragma unroll
    for (int i = 0; i < 8; ++i) z[i] = 0.f;
    return z;
}

// 16x32 bf16 operand fragment from row-major [R][stride] storage (A: rows=M;
// B: rows=N i.e. weight stored column-major [N][K]). Two 16-byte loads/lane.
static __device__ __forceinline__ bf16x16 load_frag(const bf16_t* src, int stride, int r0, int k0) {
    const int lane = threadIdx.x & 31;
    const int r  = r0 + (lane & 15);
    const int kb = k0 + ((lane >> 4) << 3);
    const bf16_t* p = src + (size_t)r * (size_t)stride + kb;
    bf16x16 f;
    *reinterpret_cast<uint4*>(&f)       = *reinterpret_cast<const uint4*>(p);
    *(reinterpret_cast<uint4*>(&f) + 1) = *reinterpret_cast<const uint4*>(p + 16);
    return f;
}

// Store a 16x16 f32 D tile into bf16 LDS, row-major [M][stride].
static __device__ __forceinline__ void store_tile(bf16_t* dst, int stride, int m0, int n0, const f32x8& acc) {
    const int lane = threadIdx.x & 31;
    const int col = n0 + (lane & 15);
    const int rb  = m0 + ((lane >> 4) << 3);
#pragma unroll
    for (int d = 0; d < 8; ++d) dst[(size_t)(rb + d) * stride + col] = (bf16_t)acc[d];
}

// Transposed store: dst[N][stride] so V becomes the column-major B operand.
static __device__ __forceinline__ void store_tile_T(bf16_t* dst, int stride, int m0, int n0, const f32x8& acc) {
    const int lane = threadIdx.x & 31;
    const int col = n0 + (lane & 15);
    const int rb  = m0 + ((lane >> 4) << 3);
#pragma unroll
    for (int d = 0; d < 8; ++d) dst[(size_t)col * stride + (rb + d)] = (bf16_t)acc[d];
}

// LayerNorm over C=256 for 128 tokens; input sXT[c][132] (transposed), output
// sH[token][264] bf16.
static __device__ __forceinline__ void layernorm_128x256(const float* sXT, bf16_t* sH,
                                                         float* sMu, float* sRs,
                                                         const float* __restrict__ g,
                                                         const float* __restrict__ b, int tid) {
    if (tid < 128) {
        float s = 0.f, s2 = 0.f;
#pragma unroll 4
        for (int c = 0; c < 256; ++c) { float v = sXT[c * 132 + tid]; s += v; s2 += v * v; }
        float mu  = s * (1.f / 256.f);
        float var = s2 * (1.f / 256.f) - mu * mu;
        sMu[tid] = mu;
        sRs[tid] = rsqrtf(var + 1e-5f);
    }
    __syncthreads();
    const float gc = g[tid];   // tid = channel
    const float bc = b[tid];
    for (int t = 0; t < 128; ++t) {
        float v = (sXT[tid * 132 + t] - sMu[t]) * sRs[t] * gc + bc;
        sH[t * 264 + tid] = (bf16_t)v;
    }
    __syncthreads();
}

// Tensor Data Mover: DMA the 16x8x256 window tile (f32) from global into LDS.
// Tile order X(w=16),Y(h=8,stride 256),Z(c=256,stride 65536); pad 4 DWORDs
// every 128 DWORDs -> LDS layout sXT[c][132] with 16B-aligned rows.
static __device__ __forceinline__ void tdm_window_gather(const float* x, int b, int h0, int w0, void* ldsDst) {
    const uint64_t ga = (uint64_t)(uintptr_t)x +
                        ((uint64_t)b * 16777216ull + (uint64_t)h0 * 256ull + (uint64_t)w0) * 4ull;
    const uint32_t lds = (uint32_t)(uintptr_t)ldsDst;

    u32x4 g0;
    g0[0] = 1u;                                                  // count=1 (valid user D#)
    g0[1] = lds;                                                 // lds_addr (bytes)
    g0[2] = (uint32_t)(ga & 0xffffffffu);                        // global_addr[31:0]
    g0[3] = (uint32_t)((ga >> 32) & 0x01ffffffu) | 0x80000000u;  // global_addr[56:32] | type=2

    i32x8 g1;
    g1[0] = 0x07920000;                 // data_size=4B | pad_enable | pad_interval=128DW | pad_amount=4DW
    g1[1] = (int)(256u << 16);          // tensor_dim0 = 256        (bits 79:48, low part)
    g1[2] = (int)(256u << 16);          // dim0 hi=0 | tensor_dim1 = 256 (low)
    g1[3] = (int)(16u << 16);           // dim1 hi=0 | tile_dim0 = 16
    g1[4] = (int)(8u | (256u << 16));   // tile_dim1 = 8 | tile_dim2 = 256
    g1[5] = 256;                        // tensor_dim0_stride = 256
    g1[6] = 0;                          // stride0 hi | stride1 lo (65536 & 0xffff = 0)
    g1[7] = 1;                          // stride1 hi (65536 >> 16)

    i32x4 g2;
    g2[0] = 256;                        // tensor_dim2
    g2[1] = 0;                          // tensor_dim3 (unused)
    g2[2] = 65536;                      // tensor_dim2_stride
    g2[3] = 0;                          // stride2 hi | tile_dim3 = 0

    i32x4 g3;
    g3[0] = 0; g3[1] = 0; g3[2] = 0; g3[3] = 0;

#if __clang_major__ >= 23
    i32x8 g4;
#pragma unroll
    for (int i = 0; i < 8; ++i) g4[i] = 0;
    __builtin_amdgcn_tensor_load_to_lds(g0, g1, g2, g3, g4, 0);
#else
    __builtin_amdgcn_tensor_load_to_lds(g0, g1, g2, g3, 0);
#endif
}

// Convert + transpose all weights to bf16 [N][K] once (then L2-resident).
__global__ void prep_weights(const float* __restrict__ wqkv, const float* __restrict__ wproj,
                             const float* __restrict__ wfc1, const float* __restrict__ wfc2,
                             bf16_t* __restrict__ ws) {
    bf16_t* q  = ws;                   // [768][256]
    bf16_t* p  = q  + 768 * 256;       // [256][256]
    bf16_t* f1 = p  + 256 * 256;       // [1024][256]
    bf16_t* f2 = f1 + 1024 * 256;      // [256][1024]
    const int step = gridDim.x * blockDim.x;
    const int t0   = blockIdx.x * blockDim.x + threadIdx.x;
    for (int i = t0; i < 768 * 256;  i += step) { int n = i >> 8,  k = i & 255;  q[i]  = (bf16_t)wqkv[k * 768 + n]; }
    for (int i = t0; i < 256 * 256;  i += step) { int n = i >> 8,  k = i & 255;  p[i]  = (bf16_t)wproj[k * 256 + n]; }
    for (int i = t0; i < 1024 * 256; i += step) { int n = i >> 8,  k = i & 255;  f1[i] = (bf16_t)wfc1[k * 1024 + n]; }
    for (int i = t0; i < 1024 * 256; i += step) { int n = i >> 10, k = i & 1023; f2[i] = (bf16_t)wfc2[k * 256 + n]; }
}

__global__ __launch_bounds__(256, 1)
void swin_block(const float* __restrict__ x,
                const float* __restrict__ b_proj,
                const float* __restrict__ g1, const float* __restrict__ b1,
                const float* __restrict__ g2, const float* __restrict__ b2,
                const float* __restrict__ b_fc1, const float* __restrict__ b_fc2,
                const bf16_t* __restrict__ wqkv_t, const bf16_t* __restrict__ wproj_t,
                const bf16_t* __restrict__ wfc1_t, const bf16_t* __restrict__ wfc2_t,
                float* __restrict__ out) {
    // ---- LDS (~315 KB of the 320 KB CDNA5 WGP budget) ----
    __shared__ alignas(16) float  sXT[256 * 132];  // f32 residual stream, [channel][token]
    __shared__ alignas(16) bf16_t sH[128 * 264];   // LN output = GEMM A operand
    __shared__ alignas(16) bf16_t sO[128 * 264];   // attn concat output / GELU buffer
    __shared__ alignas(16) bf16_t sVt[32 * 136];   // per-head V^T  [dim][token]
    __shared__ alignas(16) bf16_t sScr[17408];     // union: (sQ | sK) or sP
    __shared__ float sMu[128], sRs[128];

    bf16_t* sQ = sScr;                 // [128][40]
    bf16_t* sK = sScr + 128 * 40;      // [128][40]  (token-major == K^T column-major)
    bf16_t* sP = sScr;                 // [128][136] softmax probs

    const int tid  = threadIdx.x;
    const int wave = tid >> 5;
    const int m0   = wave << 4;        // this wave's 16-row block
    const int lane = tid & 31;

    const int wid = blockIdx.x;        // 2048 windows: b, hblock(32), wblock(16)
    const int b   = wid >> 9;
    const int rem = wid & 511;
    const int h0  = (rem >> 4) << 3;
    const int w0  = (rem & 15) << 4;

    // ---- TDM gather: sXT[c][token] = x[b][c][h0+th][w0+tw] ----
    if (wave == 0) {
        tdm_window_gather(x, b, h0, w0, sXT);
        __builtin_amdgcn_s_wait_tensorcnt(0);
    }
    __syncthreads();

    // ---- LN1 -> sH (bf16) ----
    layernorm_128x256(sXT, sH, sMu, sRs, g1, b1, tid);

    // ---- attention, head by head ----
    for (int h = 0; h < 8; ++h) {
        // QKV GEMM for this head: rows m0..m0+15, 6 column tiles (q0,q1,k0,k1,v0,v1)
        const int cols[6] = { h * 32, h * 32 + 16, 256 + h * 32, 256 + h * 32 + 16,
                              512 + h * 32, 512 + h * 32 + 16 };
#pragma unroll
        for (int t6 = 0; t6 < 6; ++t6) {
            f32x8 acc = zero8();
#pragma unroll
            for (int kk = 0; kk < 8; ++kk) {
                bf16x16 a  = load_frag(sH, 264, m0, kk * 32);
                bf16x16 bb = load_frag(wqkv_t, 256, cols[t6], kk * 32);
                acc = wmma_bf16(a, bb, acc);
            }
            const int nloc = (t6 & 1) * 16;
            if (t6 < 2)      store_tile(sQ, 40, m0, nloc, acc);
            else if (t6 < 4) store_tile(sK, 40, m0, nloc, acc);
            else             store_tile_T(sVt, 136, m0, nloc, acc);  // V^T for P@V
        }
        __syncthreads();

        // S = Q K^T (K=32 -> one WMMA per 16x16 tile); wave owns its 16 rows x 128 cols
        f32x8 s[8];
        {
            bf16x16 a0 = load_frag(sQ, 40, m0, 0);
#pragma unroll
            for (int t = 0; t < 8; ++t) {
                bf16x16 bb = load_frag(sK, 40, t * 16, 0);
                s[t] = wmma_bf16(a0, bb, zero8());
            }
        }
        __syncthreads();   // all waves done reading sQ/sK before sP overwrites them

        // softmax in registers on the C-layout; rows = (d, lane-half)
        const float scale = 0.17677669529663687f;  // 1/sqrt(32)
#pragma unroll
        for (int d = 0; d < 8; ++d) {
            float mx = -3.4e38f;
#pragma unroll
            for (int t = 0; t < 8; ++t) mx = fmaxf(mx, s[t][d]);
            mx = fmaxf(mx, __shfl_xor(mx, 1, 32));
            mx = fmaxf(mx, __shfl_xor(mx, 2, 32));
            mx = fmaxf(mx, __shfl_xor(mx, 4, 32));
            mx = fmaxf(mx, __shfl_xor(mx, 8, 32));
            float e[8], sum = 0.f;
#pragma unroll
            for (int t = 0; t < 8; ++t) { e[t] = __expf((s[t][d] - mx) * scale); sum += e[t]; }
            sum += __shfl_xor(sum, 1, 32);
            sum += __shfl_xor(sum, 2, 32);
            sum += __shfl_xor(sum, 4, 32);
            sum += __shfl_xor(sum, 8, 32);
            const float inv = 1.f / sum;
            const int row = m0 + d + ((lane >> 4) << 3);
            const int col = lane & 15;
#pragma unroll
            for (int t = 0; t < 8; ++t) sP[row * 136 + t * 16 + col] = (bf16_t)(e[t] * inv);
        }

        // O_h = P @ V  (A = own sP rows, B = sVt column-major) -> sO concat slot
#pragma unroll
        for (int ct = 0; ct < 2; ++ct) {
            f32x8 acc = zero8();
#pragma unroll
            for (int kk = 0; kk < 4; ++kk) {
                bf16x16 a  = load_frag(sP, 136, m0, kk * 32);
                bf16x16 bb = load_frag(sVt, 136, ct * 16, kk * 32);
                acc = wmma_bf16(a, bb, acc);
            }
            store_tile(sO, 264, m0, h * 32 + ct * 16, acc);
        }
        __syncthreads();   // protect sVt/sScr before next head
    }

    // ---- proj + bias + residual into sXT ----
    for (int ct = 0; ct < 16; ++ct) {
        f32x8 acc = zero8();
#pragma unroll
        for (int kk = 0; kk < 8; ++kk) {
            bf16x16 a  = load_frag(sO, 264, m0, kk * 32);
            bf16x16 bb = load_frag(wproj_t, 256, ct * 16, kk * 32);
            acc = wmma_bf16(a, bb, acc);
        }
        const int col = ct * 16 + (lane & 15);   // channel
        const int rb  = m0 + ((lane >> 4) << 3); // token
        const float bp = b_proj[col];
#pragma unroll
        for (int d = 0; d < 8; ++d) sXT[col * 132 + rb + d] += acc[d] + bp;
    }
    __syncthreads();

    // ---- LN2 -> sH ----
    layernorm_128x256(sXT, sH, sMu, sRs, g2, b2, tid);

    // ---- MLP: fc1 (4 chunks of 256 cols) + exact GELU -> sO, fc2 acc in regs ----
    f32x8 accF[16];
#pragma unroll
    for (int ct = 0; ct < 16; ++ct) accF[ct] = zero8();

    for (int cc = 0; cc < 4; ++cc) {
        for (int ct = 0; ct < 16; ++ct) {
            f32x8 acc = zero8();
#pragma unroll
            for (int kk = 0; kk < 8; ++kk) {
                bf16x16 a  = load_frag(sH, 264, m0, kk * 32);
                bf16x16 bb = load_frag(wfc1_t, 256, cc * 256 + ct * 16, kk * 32);
                acc = wmma_bf16(a, bb, acc);
            }
            const int col = cc * 256 + ct * 16 + (lane & 15);
            const int rb  = m0 + ((lane >> 4) << 3);
            const float bf1 = b_fc1[col];
#pragma unroll
            for (int d = 0; d < 8; ++d) {
                float v = acc[d] + bf1;
                float gl = 0.5f * v * (1.f + erff(v * 0.70710678118654752f));
                sO[(size_t)(rb + d) * 264 + ct * 16 + (lane & 15)] = (bf16_t)gl;
            }
        }
        for (int ct = 0; ct < 16; ++ct) {
#pragma unroll
            for (int kk = 0; kk < 8; ++kk) {
                bf16x16 a  = load_frag(sO, 264, m0, kk * 32);
                bf16x16 bb = load_frag(wfc2_t, 1024, ct * 16, cc * 256 + kk * 32);
                accF[ct] = wmma_bf16(a, bb, accF[ct]);
            }
        }
    }
    // fc2 bias + residual
    for (int ct = 0; ct < 16; ++ct) {
        const int col = ct * 16 + (lane & 15);
        const int rb  = m0 + ((lane >> 4) << 3);
        const float bf2 = b_fc2[col];
#pragma unroll
        for (int d = 0; d < 8; ++d) sXT[col * 132 + rb + d] += accF[ct][d] + bf2;
    }
    __syncthreads();

    // ---- vectorized scatter: out[b][c][h0+th][w0+tw] = sXT[c][token] ----
    {
        const size_t base = (size_t)b * 256 * 65536;
        for (int i = tid; i < 8192; i += 256) {
            int c = i >> 5, rr = i & 31;
            int th = rr >> 2, q = (rr & 3) << 2;    // token run of 4, 16B aligned
            const float4 v = *reinterpret_cast<const float4*>(&sXT[c * 132 + th * 16 + q]);
            *reinterpret_cast<float4*>(&out[base + (size_t)c * 65536 + (size_t)(h0 + th) * 256 + (w0 + q)]) = v;
        }
    }
}

extern "C" void kernel_launch(void* const* d_in, const int* in_sizes, int n_in,
                              void* d_out, int out_size, void* d_ws, size_t ws_size,
                              hipStream_t stream) {
    (void)in_sizes; (void)n_in; (void)out_size; (void)ws_size;
    const float* x     = (const float*)d_in[0];
    const float* wqkv  = (const float*)d_in[1];
    const float* wproj = (const float*)d_in[2];
    const float* bproj = (const float*)d_in[3];
    const float* g1    = (const float*)d_in[4];
    const float* b1    = (const float*)d_in[5];
    const float* g2    = (const float*)d_in[6];
    const float* b2    = (const float*)d_in[7];
    const float* wfc1  = (const float*)d_in[8];
    const float* bfc1  = (const float*)d_in[9];
    const float* wfc2  = (const float*)d_in[10];
    const float* bfc2  = (const float*)d_in[11];

    bf16_t* ws      = (bf16_t*)d_ws;              // needs 1.5 MB of workspace
    bf16_t* wqkv_t  = ws;
    bf16_t* wproj_t = wqkv_t  + 768 * 256;
    bf16_t* wfc1_t  = wproj_t + 256 * 256;
    bf16_t* wfc2_t  = wfc1_t  + 1024 * 256;

    prep_weights<<<512, 256, 0, stream>>>(wqkv, wproj, wfc1, wfc2, ws);
    swin_block<<<2048, 256, 0, stream>>>(x, bproj, g1, b1, g2, b2, bfc1, bfc2,
                                         wqkv_t, wproj_t, wfc1_t, wfc2_t, (float*)d_out);
}
